// Model_0_26972394619433
// MI455X (gfx1250) — matrix operand order repaired and checked by reading, not compiled, not run on hardware
//
#include <hip/hip_runtime.h>

typedef __attribute__((ext_vector_type(16))) _Float16 v16h;
typedef __attribute__((ext_vector_type(8)))  float    v8f;

typedef unsigned int tdm_u32x4 __attribute__((ext_vector_type(4)));
typedef int          tdm_i32x4 __attribute__((ext_vector_type(4)));
typedef int          tdm_i32x8 __attribute__((ext_vector_type(8)));

#ifndef __has_builtin
#define __has_builtin(x) 0
#endif
#if __has_builtin(__builtin_amdgcn_tensor_load_to_lds) && \
    __has_builtin(__builtin_amdgcn_s_wait_tensorcnt)
#define USE_TDM 1
#else
#define USE_TDM 0
#endif

#if USE_TDM
// ---------------------------------------------------------------------------
// Contiguous global->LDS DMA via the Tensor Data Mover.
// D# per CDNA5 ISA: one row of n 2-byte elements
// (count=1, type=2 "image", data_size=2B, tile_dim0=n, tensor_dim1=tile_dim1=1,
// no padding/iterate/gather, workgroup_mask=0 -> non-cluster).
// Issue once per workgroup (wave 0); TDM ignores EXEC. Tracked by TENSORcnt.
// ---------------------------------------------------------------------------
__device__ __forceinline__ void tdm_load_1d(void* lds_dst, const void* gsrc,
                                            unsigned n_elems_2b) {
  unsigned lds_addr = (unsigned)(size_t)lds_dst;          // low 32 bits = LDS offset
  unsigned long long ga = (unsigned long long)(size_t)gsrc;
  tdm_u32x4 g0;
  g0[0] = 1u;                                             // count=1, user D#
  g0[1] = lds_addr;                                       // lds_addr [63:32]
  g0[2] = (unsigned)(ga & 0xffffffffu);                   // global_addr[31:0]
  g0[3] = (unsigned)((ga >> 32) & 0x01ffffffu) | (2u << 30); // addr[56:32], type=2
  tdm_i32x8 g1;
  g1[0] = (int)(1u << 16);                                // data_size=1 (2 bytes)
  g1[1] = (int)((n_elems_2b & 0xffffu) << 16);            // tensor_dim0[15:0]
  g1[2] = (int)((n_elems_2b >> 16) & 0xffffu) | (1 << 16); // dim0[31:16], tensor_dim1=1
  g1[3] = (int)((n_elems_2b & 0xffffu) << 16);            // tile_dim0 = n
  g1[4] = 1;                                              // tile_dim1 = 1
  g1[5] = (int)n_elems_2b;                                // tensor_dim0_stride lo
  g1[6] = 0;
  g1[7] = 0;
  tdm_i32x4 zz = {0, 0, 0, 0};
#if __clang_major__ >= 23
  tdm_i32x8 z8 = {0, 0, 0, 0, 0, 0, 0, 0};
  __builtin_amdgcn_tensor_load_to_lds(g0, g1, zz, zz, z8, 0);
#else
  __builtin_amdgcn_tensor_load_to_lds(g0, g1, zz, zz, 0);
#endif
}
#endif

// ---------------------------------------------------------------------------
// K1: conv1 (1->10ch, 3x3 SAME) + spike + 2x2 maxpool, fused.
// Output p1: [B][14][14][10] f16, values in {0,1}.
// ---------------------------------------------------------------------------
__global__ __launch_bounds__(256)
void conv1_pool_spike(const float* __restrict__ x,
                      const float* __restrict__ Wc1,
                      _Float16* __restrict__ p1) {
  int idx = blockIdx.x * blockDim.x + threadIdx.x;
  if (idx >= 4096 * 10 * 196) return;
  int b   = idx / (10 * 196);
  int rem = idx - b * 10 * 196;
  int c   = rem / 196;
  int p   = rem - c * 196;
  int py = p / 14, px = p - py * 14;

  float w[9];
#pragma unroll
  for (int i = 0; i < 9; ++i) w[i] = Wc1[c * 9 + i];

  const float* xi = x + (size_t)b * 784;
  float best = 0.0f;                  // spikes are >= 0, so 0-init == maxpool
#pragma unroll
  for (int dy = 0; dy < 2; ++dy) {
#pragma unroll
    for (int dx = 0; dx < 2; ++dx) {
      int oy = 2 * py + dy, ox = 2 * px + dx;
      float acc = 0.0f;
#pragma unroll
      for (int ky = 0; ky < 3; ++ky) {
        int iy = oy + ky - 1;
#pragma unroll
        for (int kx = 0; kx < 3; ++kx) {
          int ix = ox + kx - 1;
          float v = (iy >= 0 && iy < 28 && ix >= 0 && ix < 28) ? xi[iy * 28 + ix] : 0.0f;
          acc = fmaf(v, w[ky * 3 + kx], acc);
        }
      }
      float s = (acc - 1.0f >= 0.0f) ? 1.0f : 0.0f;   // spike(conv - V_TH)
      best = fmaxf(best, s);
    }
  }
  p1[(size_t)(b * 196 + p) * 10 + c] = (_Float16)best;
}

// ---------------------------------------------------------------------------
// K2: conv2 (10->20ch, 3x3 SAME on 14x14) as implicit GEMM via WMMA,
//     fused spike + 2x2 maxpool. One workgroup (8 waves) per image.
// GEMM: M=196 rows (pixels), K=90 (pad 96), N=20 (pad 32).
// p1 image staged to LDS via TDM; z: [B][1024] f16 (cols 980..1023 zeroed).
// ---------------------------------------------------------------------------
__global__ __launch_bounds__(256)
void conv2_pool_spike(const _Float16* __restrict__ p1,
                      const float* __restrict__ Wc2,
                      _Float16* __restrict__ z) {
  __shared__ __align__(16) _Float16 p1s[1960];      // 14*14*10 image
  __shared__ __align__(16) _Float16 s2s[208 * 32];  // spikes [row][ch]

  int tid = threadIdx.x;
  int b   = blockIdx.x;

  int lane = tid & 31;
  int wave = tid >> 5;
  int lmod = lane & 15;
  int hi   = lane >> 4;   // 0: low half-wave, 1: high half-wave

  // stage p1 image into LDS
#if USE_TDM
  if (wave == 0) {
    tdm_load_1d(p1s, p1 + (size_t)b * 1960, 1960);
    __builtin_amdgcn_s_wait_tensorcnt(0);
  }
#else
  {
    const unsigned int* src = (const unsigned int*)(p1 + (size_t)b * 1960);
    unsigned int* dst = (unsigned int*)p1s;
    for (int i = tid; i < 980; i += 256) dst[i] = src[i];
  }
#endif

  // Build B operand (Wc2) in registers, ISA 16-bit B layout:
  //   lane -> n = ntile*16 + (lane&15);  k = kc*32 + {0..7,16..23}[e] + hi*8
  // K order: k = (ky*3+kx)*10 + ci  (matches A-patch gather below)
  v16h bm[3][2];
#pragma unroll
  for (int kc = 0; kc < 3; ++kc) {
#pragma unroll
    for (int nt = 0; nt < 2; ++nt) {
      v16h bb;
#pragma unroll
      for (int e = 0; e < 16; ++e) {
        int k = kc * 32 + ((e < 8) ? e : (16 + (e - 8))) + hi * 8;
        int n = nt * 16 + lmod;
        float v = 0.0f;
        if (k < 90 && n < 20) {
          int kk = k / 10, ci = k - kk * 10;      // kk = ky*3+kx
          v = Wc2[n * 90 + ci * 9 + kk];
        }
        bb[e] = (_Float16)v;
      }
      bm[kc][nt] = bb;
    }
  }
  __syncthreads();

  // 13 M-tiles of 16 rows, round-robin over 8 waves
  for (int mt = wave; mt < 13; mt += 8) {
    int r = mt * 16 + lmod;                 // A row for this lane
    int valid = (r < 196);
    int y = r / 14, xx = r - y * 14;
    v8f acc0 = {};
    v8f acc1 = {};
#pragma unroll
    for (int kc = 0; kc < 3; ++kc) {
      // A layout (16-bit 16x32): lane<16: k = {0..7,16..23}; lane>=16: +8
      v16h a;
#pragma unroll
      for (int e = 0; e < 16; ++e) {
        int k = kc * 32 + ((e < 8) ? e : (16 + (e - 8))) + hi * 8;
        _Float16 v = (_Float16)0.0f;
        if (valid && k < 90) {
          int kk = k / 10, ci = k - kk * 10;
          int ky = kk / 3, kx = kk - ky * 3;
          int iy = y + ky - 1, ix = xx + kx - 1;
          if (iy >= 0 && iy < 14 && ix >= 0 && ix < 14)
            v = p1s[(iy * 14 + ix) * 10 + ci];
        }
        a[e] = v;
      }
      acc0 = __builtin_amdgcn_wmma_f32_16x16x32_f16(false, a, false, bm[kc][0],
                                                    (short)0, acc0, false, false);
      acc1 = __builtin_amdgcn_wmma_f32_16x16x32_f16(false, a, false, bm[kc][1],
                                                    (short)0, acc1, false, false);
    }
    // C layout: VGPR rr -> M = rr + 8*hi, N = lmod; spike and park in LDS
#pragma unroll
    for (int rr = 0; rr < 8; ++rr) {
      int row = mt * 16 + rr + hi * 8;
      s2s[row * 32 + lmod]      = (_Float16)((acc0[rr] >= 1.0f) ? 1.0f : 0.0f);
      s2s[row * 32 + 16 + lmod] = (_Float16)((acc1[rr] >= 1.0f) ? 1.0f : 0.0f);
    }
  }
  __syncthreads();

  // maxpool 2x2 -> z[b][c*49 + py*7 + px] (CHW flatten, K padded to 1024)
  _Float16* zb = z + (size_t)b * 1024;
  for (int q = tid; q < 1024; q += 256) {
    _Float16 v = (_Float16)0.0f;
    if (q < 980) {
      int c   = q / 49;
      int rem = q - c * 49;
      int py = rem / 7, px = rem - py * 7;
      int r00 = (2 * py) * 14 + 2 * px;
      _Float16 a0 = s2s[r00 * 32 + c];
      _Float16 a1 = s2s[(r00 + 1) * 32 + c];
      _Float16 a2 = s2s[(r00 + 14) * 32 + c];
      _Float16 a3 = s2s[(r00 + 15) * 32 + c];
      _Float16 m01 = (a0 > a1) ? a0 : a1;
      _Float16 m23 = (a2 > a3) ? a2 : a3;
      v = (m01 > m23) ? m01 : m23;
    }
    zb[q] = v;
  }
}

// ---------------------------------------------------------------------------
// K3: fc1  h[4096][100] = z[4096][980] @ Wf1^T  via WMMA.
// M=4096 (blocks of 16), K=1024 (zero-padded), N=128 (100 padded, 8 waves).
// A block (32 KB contiguous) staged to LDS via one TDM descriptor.
// h stored [4096][128] f32 (cols >=100 are exact zeros from padded B).
// ---------------------------------------------------------------------------
__global__ __launch_bounds__(256)
void fc1_wmma(const _Float16* __restrict__ z,
              const float* __restrict__ Wf1,
              float* __restrict__ h) {
  __shared__ __align__(16) _Float16 as[16 * 1024];   // 32 KB A block
  int tid  = threadIdx.x;
  int mblk = blockIdx.x;    // 0..255

  int lane = tid & 31, wave = tid >> 5;
  int lmod = lane & 15, hi = lane >> 4;
  int nbase = wave * 16;

#if USE_TDM
  if (wave == 0) {
    tdm_load_1d(as, z + (size_t)mblk * 16 * 1024, 16 * 1024);
    __builtin_amdgcn_s_wait_tensorcnt(0);
  }
#else
  {
    const unsigned int* src = (const unsigned int*)(z + (size_t)mblk * 16 * 1024);
    unsigned int* dst = (unsigned int*)as;
    for (int i = tid; i < 8192; i += 256) dst[i] = src[i];
  }
#endif

  __builtin_prefetch(Wf1 + (size_t)nbase * 980, 0, 0);
  __syncthreads();

  v8f acc = {};
  for (int kc = 0; kc < 32; ++kc) {
    v16h a, bb;
#pragma unroll
    for (int e = 0; e < 16; ++e) {
      int ka = kc * 32 + ((e < 8) ? e : (16 + (e - 8))) + hi * 8;
      a[e] = as[lmod * 1024 + ka];
      int kb = kc * 32 + ((e < 8) ? e : (16 + (e - 8))) + hi * 8;
      int n  = nbase + lmod;
      float wv = (n < 100 && kb < 980) ? Wf1[n * 980 + kb] : 0.0f;
      bb[e] = (_Float16)wv;
    }
    acc = __builtin_amdgcn_wmma_f32_16x16x32_f16(false, a, false, bb,
                                                 (short)0, acc, false, false);
  }
#pragma unroll
  for (int rr = 0; rr < 8; ++rr) {
    int row = mblk * 16 + rr + hi * 8;
    h[(size_t)row * 128 + nbase + lmod] = acc[rr];
  }
}

// ---------------------------------------------------------------------------
// K4: T=32 LIF recurrence (serial scan, register-resident state) + fc2
//     per timestep + spike counting + softmax. One thread per batch row.
// ---------------------------------------------------------------------------
__global__ __launch_bounds__(128)
void lif_loop_softmax(const float* __restrict__ h,
                      const float* __restrict__ Wf2,
                      float* __restrict__ out) {
  __shared__ float w2[1000];
  int tid = threadIdx.x;
  for (int i = tid; i < 1000; i += 128) w2[i] = Wf2[i];
  __syncthreads();

  int row = blockIdx.x * 128 + tid;
  const float* hr = h + (size_t)row * 128;

  float hreg[100];
#pragma unroll
  for (int j = 0; j < 100; ++j) hreg[j] = hr[j];
  float v1[100];
#pragma unroll
  for (int j = 0; j < 100; ++j) v1[j] = 0.0f;
  float v2[10], cnt[10];
#pragma unroll
  for (int o = 0; o < 10; ++o) { v2[o] = 0.0f; cnt[o] = 0.0f; }

  for (int t = 0; t < 32; ++t) {
    float y[10];
#pragma unroll
    for (int o = 0; o < 10; ++o) y[o] = 0.0f;
#pragma unroll
    for (int j = 0; j < 100; ++j) {
      float v  = v1[j] + (hreg[j] - v1[j]) * 0.5f;   // v += (h - v)/tau, tau=2
      float o1 = (v - 1.0f >= 0.0f) ? 1.0f : 0.0f;   // spike(v - V_TH)
      v1[j] = v * (1.0f - o1);                        // hard reset
#pragma unroll
      for (int o = 0; o < 10; ++o) y[o] = fmaf(o1, w2[o * 100 + j], y[o]);
    }
#pragma unroll
    for (int o = 0; o < 10; ++o) {
      float v  = v2[o] + (y[o] - v2[o]) * 0.5f;
      float o2 = (v - 1.0f >= 0.0f) ? 1.0f : 0.0f;
      v2[o] = v * (1.0f - o2);
      cnt[o] += o2;
    }
  }

  // softmax(cnt / 32, axis=1)
  float mx = -1e30f;
#pragma unroll
  for (int o = 0; o < 10; ++o) mx = fmaxf(mx, cnt[o] * (1.0f / 32.0f));
  float e[10];
  float s = 0.0f;
#pragma unroll
  for (int o = 0; o < 10; ++o) { e[o] = __expf(cnt[o] * (1.0f / 32.0f) - mx); s += e[o]; }
  float inv = 1.0f / s;
#pragma unroll
  for (int o = 0; o < 10; ++o) out[(size_t)row * 10 + o] = e[o] * inv;
}

// ---------------------------------------------------------------------------
extern "C" void kernel_launch(void* const* d_in, const int* in_sizes, int n_in,
                              void* d_out, int out_size, void* d_ws, size_t ws_size,
                              hipStream_t stream) {
  (void)in_sizes; (void)n_in; (void)out_size; (void)ws_size;
  const float* x   = (const float*)d_in[0];
  const float* Wc1 = (const float*)d_in[1];
  const float* Wc2 = (const float*)d_in[2];
  const float* Wf1 = (const float*)d_in[3];
  const float* Wf2 = (const float*)d_in[4];
  float* out = (float*)d_out;

  char* ws = (char*)d_ws;
  size_t off = 0;
  _Float16* p1 = (_Float16*)(ws + off);
  off += (size_t)4096 * 1960 * sizeof(_Float16);          // 16,056,320 B (256-aligned)
  _Float16* z = (_Float16*)(ws + off);
  off += (size_t)4096 * 1024 * sizeof(_Float16);          // 8,388,608 B
  float* h = (float*)(ws + off);                          // 2,097,152 B

  conv1_pool_spike<<<31360, 256, 0, stream>>>(x, Wc1, p1);
  conv2_pool_spike<<<4096, 256, 0, stream>>>(p1, Wc2, z);
  fc1_wmma<<<256, 256, 0, stream>>>(z, Wf1, h);
  lif_loop_softmax<<<32, 128, 0, stream>>>(h, Wf2, out);
}
